// LSTM_8985071583782
// MI455X (gfx1250) — compile-verified, matrix operand order checked by reading, not run on hardware
//
#include <hip/hip_runtime.h>
#include <hip/hip_bf16.h>

// ---------------- problem constants ----------------
#define T_STEPS 1024
#define BATCH   64
#define ISZ     256
#define HSZ     512
#define G4H     2048   // 4*H
#define NBLK2   16     // persistent blocks in the recurrent kernel

typedef __bf16 v8bf  __attribute__((ext_vector_type(8)));
typedef __bf16 v16bf __attribute__((ext_vector_type(16)));
typedef float  v8f   __attribute__((ext_vector_type(8)));

// ---------------------------------------------------------------------------
// A-fragment loader: 16-bit A 16x32 layout. Lane l (m = l&15, grp = l>>4)
// holds K = [kt*32 + 8*grp .. +7] and [kt*32 + 16 + 8*grp .. +7]: two
// contiguous 16B runs from a row-major bf16 matrix -> two global_load_b128.
// ---------------------------------------------------------------------------
__device__ inline v16bf load_a_frag(const __bf16* __restrict__ A, int mrow,
                                    int ldk, int kbase, int lane) {
    const __bf16* p = A + (long)mrow * ldk + kbase + ((lane >> 4) << 3);
    v8bf lo = *(const v8bf*)p;
    v8bf hi = *(const v8bf*)(p + 16);
    return __builtin_shufflevector(lo, hi, 0,1,2,3,4,5,6,7,8,9,10,11,12,13,14,15);
}

__device__ inline float sigmoidf_fast(float x) {
    return 1.0f / (1.0f + __expf(-x));
}

// ---------------- phase 0: convert x to bf16 ----------------
__global__ void k_cvt_x(const float* __restrict__ x, __bf16* __restrict__ xbf, long n) {
    long i = (long)blockIdx.x * 256 + threadIdx.x;
    if (i < n) xbf[i] = (__bf16)x[i];
}

// ---------------- phase 0: pack W gate slices into WMMA B-fragment order ----
// Fragment (kt, ct): 32 lanes x 16 halfs, lane-major contiguous (512 halfs).
// Lane l -> column n = ct*16 + (l&15); element e -> krel = (e&7) + 8*(l>>4) + 16*(e>>3).
__global__ void k_pack_w(const float* __restrict__ Wf, const float* __restrict__ Wi,
                         const float* __restrict__ Wc, const float* __restrict__ Wo,
                         int rowOff, int ktiles, __bf16* __restrict__ dst) {
    long i = (long)blockIdx.x * 256 + threadIdx.x;
    long total = (long)ktiles * 128 * 512;
    if (i >= total) return;
    int  e    = (int)(i & 15);
    int  lane = (int)((i >> 4) & 31);
    long frag = i >> 9;
    int  ct   = (int)(frag & 127);
    int  kt   = (int)(frag >> 7);
    int  n    = ct * 16 + (lane & 15);     // global column in [0, 2048)
    int  g    = n >> 9;                    // which gate
    int  ncol = n & 511;                   // column within gate
    int  krel = (e & 7) + 8 * (lane >> 4) + 16 * (e >> 3);
    int  row  = rowOff + kt * 32 + krel;
    const float* Wg = (g == 0) ? Wf : (g == 1) ? Wi : (g == 2) ? Wc : Wo;
    dst[i] = (__bf16)Wg[(long)row * 512 + ncol];
}

// ---------------- phase 0: zero h0 and barrier counter ----------------
__global__ void k_init(__bf16* __restrict__ h0, unsigned* __restrict__ cnt) {
    int i = blockIdx.x * 256 + threadIdx.x;
    if (i < BATCH * HSZ) h0[i] = (__bf16)0.0f;
    if (i == 0) *cnt = 0u;
}

// ---------------- phase 1: Gx = X @ Wx + bias (bf16 WMMA, f32 accum) -------
// grid = (32, 512); wave w of block computes rows [ (by*8+w)*16 , +16 ),
// columns [bx*64, +64) as 4 accumulators; K = 256 -> 8 k-steps.
__global__ void __launch_bounds__(256) k_gemm_x(const __bf16* __restrict__ xbf,
                                                const __bf16* __restrict__ wfrag,
                                                const float* __restrict__ bfv,
                                                const float* __restrict__ biv,
                                                const float* __restrict__ bcv,
                                                const float* __restrict__ bov,
                                                __bf16* __restrict__ gx) {
    int lane = threadIdx.x & 31;
    int wave = threadIdx.x >> 5;
    int mt   = blockIdx.y * 8 + wave;          // 0..4095
    int ng   = blockIdx.x;                     // 0..31 (groups of 4 n-tiles)
    int mrowA = mt * 16 + (lane & 15);

    v8f acc[4] = {};
    for (int kt = 0; kt < 8; ++kt) {
        v16bf a = load_a_frag(xbf, mrowA, ISZ, kt * 32, lane);
#pragma unroll
        for (int j = 0; j < 4; ++j) {
            int nt = ng * 4 + j;
            const __bf16* bp = wfrag + (((long)(kt * 128 + nt)) << 9) + (lane << 4);
            v16bf b = *(const v16bf*)bp;
            acc[j] = __builtin_amdgcn_wmma_f32_16x16x32_bf16(
                         false, a, false, b, (short)0, acc[j], false, false);
        }
    }
    int grp = lane >> 4;
#pragma unroll
    for (int j = 0; j < 4; ++j) {
        int nc = (ng * 4 + j) * 16 + (lane & 15);
        const float* bpt = (nc < 512) ? bfv : (nc < 1024) ? biv : (nc < 1536) ? bcv : bov;
        float bias = bpt[nc & 511];
#pragma unroll
        for (int r = 0; r < 8; ++r) {
            int m = mt * 16 + r + 8 * grp;     // C layout: VGPR r, lanes>=16 -> M+8
            gx[(long)m * G4H + nc] = (__bf16)(acc[j][r] + bias);
        }
    }
}

// ---------------- grid-wide barrier (16 resident blocks) ----------------
// Cluster split-barrier first (fast path when launched as a 16-WG cluster;
// ISA-defined NOP otherwise), then a correct L2-atomic generation barrier.
__device__ inline void grid_sync(unsigned* cnt, unsigned need) {
    __syncthreads();
    __builtin_amdgcn_s_cluster_barrier();
    if (threadIdx.x == 0) {
        __hip_atomic_fetch_add(cnt, 1u, __ATOMIC_RELEASE, __HIP_MEMORY_SCOPE_AGENT);
        while (__hip_atomic_load(cnt, __ATOMIC_ACQUIRE, __HIP_MEMORY_SCOPE_AGENT) < need) {
            __builtin_amdgcn_s_sleep(2);
        }
    }
    __syncthreads();
}

// ---------------- phase 2: persistent recurrent kernel ----------------
// 16 blocks x 256 threads. Block b owns H columns [b*32, b*32+32).
// Wave w: mt = w&3 (16 batch rows), nth = w>>2 (which 16-col subtile);
// its 4 accumulators are the 4 gates for the SAME (m,n) subtile, so the
// pointwise LSTM update is register-local. h is ping-pong double-buffered
// (single grid barrier per step); c never leaves VGPRs; Wh fragments are
// staged once into 128 KB of LDS and hit via ds_load_b128 every step.
__global__ void __launch_bounds__(256) k_lstm(const __bf16* __restrict__ whfrag,
                                              const __bf16* __restrict__ gx,
                                              __bf16* __restrict__ h0,
                                              __bf16* __restrict__ h1,
                                              const float* __restrict__ wfc,
                                              const float* __restrict__ bfc,
                                              float* __restrict__ out,
                                              unsigned* __restrict__ cnt) {
    __shared__ __bf16 sWh[16 * 8 * 512];   // [kt][j][frag 512 halfs] = 128 KB

    int lane = threadIdx.x & 31;
    int wave = threadIdx.x >> 5;
    int grp  = lane >> 4;
    int blk  = blockIdx.x;          // 0..15
    int mt   = wave & 3;            // 0..3  (covers all 64 batch rows)
    int nth  = wave >> 2;           // 0..1
    int n    = blk * 32 + nth * 16 + (lane & 15);   // this lane's H column
    int mrowA = mt * 16 + (lane & 15);

    // ---- stage this block's 128 Wh fragments into LDS (once) ----
    for (int q = threadIdx.x; q < 8192; q += 256) {
        int frag = q >> 6;              // 0..127
        int off  = (q & 63) << 3;       // 8-half chunks
        int kt   = frag >> 3;
        int j    = frag & 7;            // j = g*2 + s
        int ct   = (j >> 1) * 32 + blk * 2 + (j & 1);
        *(v8bf*)(sWh + frag * 512 + off) =
            *(const v8bf*)(whfrag + (((long)(kt * 128 + ct)) << 9) + off);
    }
    __syncthreads();

    float creg[8] = {0.f, 0.f, 0.f, 0.f, 0.f, 0.f, 0.f, 0.f};
    __bf16* hbuf[2] = { h0, h1 };

    for (int t = 0; t < T_STEPS; ++t) {
        const __bf16* hr = hbuf[t & 1];
        __bf16*       hw = hbuf[(t + 1) & 1];

        // hoisted Gx gathers (bias already folded in) — overlap with GEMM
        __bf16 gxv0[8], gxv1[8], gxv2[8], gxv3[8];
#pragma unroll
        for (int r = 0; r < 8; ++r) {
            int m = mt * 16 + r + 8 * grp;
            const __bf16* gp = gx + ((long)m * T_STEPS + t) * G4H + n;
            gxv0[r] = gp[0];
            gxv1[r] = gp[512];
            gxv2[r] = gp[1024];
            gxv3[r] = gp[1536];
            __builtin_prefetch(gp + G4H, 0, 1);   // next timestep's row
        }

        v8f acc[4] = {};
        for (int kt = 0; kt < 16; ++kt) {
            v16bf a = load_a_frag(hr, mrowA, HSZ, kt * 32, lane);
#pragma unroll
            for (int g = 0; g < 4; ++g) {
                const __bf16* bp = sWh + (((kt << 3) + (g * 2 + nth)) << 9) + (lane << 4);
                v16bf b = *(const v16bf*)bp;    // ds_load_b128 x2
                acc[g] = __builtin_amdgcn_wmma_f32_16x16x32_bf16(
                             false, a, false, b, (short)0, acc[g], false, false);
            }
        }

        // pointwise LSTM update: gates + c entirely in registers
#pragma unroll
        for (int r = 0; r < 8; ++r) {
            float gf = acc[0][r] + (float)gxv0[r];
            float gi = acc[1][r] + (float)gxv1[r];
            float gc = acc[2][r] + (float)gxv2[r];
            float go = acc[3][r] + (float)gxv3[r];
            float fg = sigmoidf_fast(gf);
            float ig = sigmoidf_fast(gi);
            float cg = tanhf(gc);
            float og = sigmoidf_fast(go);
            float cn = fg * creg[r] + ig * cg;
            creg[r] = cn;
            int m = mt * 16 + r + 8 * grp;
            hw[(long)m * HSZ + n] = (__bf16)(og * tanhf(cn));
        }

        grid_sync(cnt, (unsigned)(NBLK2 * (t + 1)));
    }

    // final: out = h @ W_fc + b_fc  (final h lives in h0 since T is even)
    if (blk == 0 && threadIdx.x < BATCH) {
        float s = bfc[0];
        const __bf16* hrow = h0 + (long)threadIdx.x * HSZ;
        for (int k = 0; k < HSZ; ++k) s += (float)hrow[k] * wfc[k];
        out[threadIdx.x] = s;
    }
}

// ---------------------------------------------------------------------------
extern "C" void kernel_launch(void* const* d_in, const int* in_sizes, int n_in,
                              void* d_out, int out_size, void* d_ws, size_t ws_size,
                              hipStream_t stream) {
    (void)in_sizes; (void)n_in; (void)out_size; (void)ws_size;
    const float* x   = (const float*)d_in[0];
    const float* Wf  = (const float*)d_in[1];
    const float* bf_ = (const float*)d_in[2];
    const float* Wi  = (const float*)d_in[3];
    const float* bi_ = (const float*)d_in[4];
    const float* Wc  = (const float*)d_in[5];
    const float* bc_ = (const float*)d_in[6];
    const float* Wo  = (const float*)d_in[7];
    const float* bo_ = (const float*)d_in[8];
    const float* Wfc = (const float*)d_in[9];
    const float* bfc = (const float*)d_in[10];
    float* out = (float*)d_out;

    char* ws = (char*)d_ws;
    size_t off = 0;
    auto alloc = [&](size_t bytes) { size_t p = off; off += (bytes + 255) & ~(size_t)255; return p; };
    __bf16*  xbf   = (__bf16*)(ws + alloc((size_t)BATCH * T_STEPS * ISZ * 2)); // 32 MB
    __bf16*  wxfrg = (__bf16*)(ws + alloc((size_t)ISZ * G4H * 2));             // 1 MB
    __bf16*  whfrg = (__bf16*)(ws + alloc((size_t)HSZ * G4H * 2));             // 2 MB
    __bf16*  gxbuf = (__bf16*)(ws + alloc((size_t)BATCH * T_STEPS * G4H * 2)); // 256 MB
    __bf16*  h0    = (__bf16*)(ws + alloc((size_t)BATCH * HSZ * 2));
    __bf16*  h1    = (__bf16*)(ws + alloc((size_t)BATCH * HSZ * 2));
    unsigned* cnt  = (unsigned*)(ws + alloc(256));

    long nx = (long)BATCH * T_STEPS * ISZ;
    k_cvt_x<<<(unsigned)((nx + 255) / 256), 256, 0, stream>>>(x, xbf, nx);
    k_pack_w<<<( 8 * 128 * 512 + 255) / 256, 256, 0, stream>>>(Wf, Wi, Wc, Wo,   0,  8, wxfrg);
    k_pack_w<<<(16 * 128 * 512 + 255) / 256, 256, 0, stream>>>(Wf, Wi, Wc, Wo, 256, 16, whfrg);
    k_init<<<(BATCH * HSZ + 255) / 256, 256, 0, stream>>>(h0, cnt);

    dim3 g1(32, 512);  // 2048 cols / 64 per block-x ; 65536 rows / (16*8 waves)
    k_gemm_x<<<g1, 256, 0, stream>>>(xbf, wxfrg, bf_, bi_, bc_, bo_, gxbuf);

    k_lstm<<<NBLK2, 256, 0, stream>>>(whfrg, gxbuf, h0, h1, Wfc, bfc, out, cnt);
}
